// MyConv3D_4698694222450
// MI455X (gfx1250) — matrix operand order repaired and checked by reading, not compiled, not run on hardware
//
#include <hip/hip_runtime.h>
#include <hip/hip_bf16.h>

typedef __attribute__((ext_vector_type(16))) __bf16 v16bf;
typedef __attribute__((ext_vector_type(8)))  float  v8f;

#define BB   4
#define CIN  16
#define COUT 64
#define DDIM 32
#define HH   64
#define WW   64
#define KVOL 27            // 3*3*3
#define KROW 432           // CIN*KVOL

// -------- ws layout --------
// [0     ..   256) : inv_norm, f32[64]
// [256   ..  8448) : dfrag,  bf16 [kc2][mi4][lane32][16]   (Gram matrix, A-frag order)
// [8448  .. 82176) : wfragA, bf16 [p9][kc2][mi4][lane32][16] (weights, A-frag order)
// ---------------------------

// A-fragment K mapping (16-bit A 16x32, contiguous order):
// lane<16: elem i -> K = i; lane>=16: +16.
__device__ __forceinline__ int a_klocal(int i, int hi) {
    return i + (hi ? 16 : 0);
}

__global__ __launch_bounds__(256) void prep_kernel(
    const float* __restrict__ w, float* __restrict__ inv_norm,
    __bf16* __restrict__ dfrag, __bf16* __restrict__ wfragA)
{
    const int gtid   = blockIdx.x * 256 + threadIdx.x;
    const int nthr   = gridDim.x * 256;

    // inv_norm (f32 precision)
    for (int c = gtid; c < COUT; c += nthr) {
        const float* wc = w + (size_t)c * KROW;
        float s = 0.0f;
        #pragma unroll 4
        for (int k = 0; k < KROW; ++k) s += wc[k] * wc[k];
        inv_norm[c] = 1.0f / (s + 1e-8f);
    }

    // Gram matrix directly in A-fragment order: dfrag[kc][mi][lane][i]
    for (int s = gtid; s < 2 * 4 * 32 * 16; s += nthr) {
        const int i    = s & 15;
        const int lane = (s >> 4) & 31;
        const int mi   = (s >> 9) & 3;
        const int kc   = (s >> 11) & 1;
        const int hi   = lane >> 4;
        const int co   = mi * 16 + (lane & 15);
        const int kg   = kc * 32 + a_klocal(i, hi);      // column of d = channel
        const float* wi = w + (size_t)co * KROW;
        const float* wj = w + (size_t)kg * KROW;
        float acc = 0.0f;
        #pragma unroll 4
        for (int k = 0; k < KROW; ++k) acc += wi[k] * wj[k];
        dfrag[s] = (__bf16)acc;
    }

    // Weights in A-fragment order: wfragA[p][kc][mi][lane][i], K = kw*16 + ci (48 real, pad 64)
    for (int s = gtid; s < 9 * 2 * 4 * 32 * 16; s += nthr) {
        const int i    = s & 15;
        const int lane = (s >> 4) & 31;
        const int mi   = (s >> 9) & 3;
        const int kc   = (s >> 11) & 1;
        const int p    = s >> 12;                        // kd*3 + kh
        const int hi   = lane >> 4;
        const int co   = mi * 16 + (lane & 15);
        const int k    = kc * 32 + a_klocal(i, hi);
        __bf16 v = (__bf16)0.0f;
        if (k < 48) {
            const int kw = k >> 4, ci = k & 15;
            const int kd = p / 3,  kh = p % 3;
            v = (__bf16)w[(size_t)co * KROW + ci * KVOL + kd * 9 + kh * 3 + kw];
        }
        wfragA[s] = v;
    }
}

// Implicit-GEMM 3D conv. Block = (b, d, h) row. x slab async-DMA'd to LDS (f32),
// transpose-converted once to bf16 [plane][w'][ci] with zero slots for halos/pad,
// then every A/B fragment is a contiguous vector load. 36 WMMAs per wave.
__global__ __launch_bounds__(256) void conv3d_kernel(
    const float* __restrict__ x, const __bf16* __restrict__ wfragA,
    float* __restrict__ out)
{
    __shared__ float  xsf[9][CIN][WW];        // raw f32 slab, filled by async DMA
    __shared__ __bf16 xsbT[9][68][CIN];       // bf16 transposed; w' slots 0..65 (64,65 = zero)

    const int blk = blockIdx.x;
    const int h   = blk & 63;
    const int dd  = (blk >> 6) & 31;
    const int b   = blk >> 11;

    const int wave   = threadIdx.x >> 5;
    const int lane   = threadIdx.x & 31;
    const int laneLo = lane & 15;
    const int hi     = lane >> 4;

    const int mi = wave >> 1;
    const int nj = (wave & 1) * 2;
    const int n0 = nj * 16 + laneLo;
    const int n1 = n0 + 16;

    // ---- async DMA: 144 rows of 256B (valid planes only), 2 rows per issue ----
    for (int pr = wave; pr < 72; pr += 8) {
        const int p  = pr >> 3;                       // plane = kd*3+kh (pairs never straddle)
        const int kd = p / 3, kh = p % 3;
        const int zd = dd + kd - 1, zh = h + kh - 1;
        if ((unsigned)zd >= DDIM || (unsigned)zh >= HH) continue;   // wave-uniform
        const int ci = ((pr & 7) << 1) | hi;          // two rows per wave-issue
        const size_t gbyte = ((((size_t)b * CIN + ci) * DDIM + zd) * HH + zh) * (WW * 4)
                           + (size_t)laneLo * 16;
        const unsigned long long ga = (unsigned long long)(uintptr_t)x + gbyte;
        const unsigned ldsoff = (unsigned)(uintptr_t)&xsf[p][ci][0] + (unsigned)(laneLo * 16);
        asm volatile("global_load_async_to_lds_b128 %0, %1, off"
                     :: "v"(ldsoff), "v"(ga) : "memory");
    }
    asm volatile("s_wait_asynccnt 0x0" ::: "memory");
    __syncthreads();

    // ---- transpose + f32->bf16, zero-fill halos and OOB planes ----
    for (int s = threadIdx.x; s < 9 * 66 * CIN; s += 256) {
        const int ci = s & 15;
        const int t  = s >> 4;
        const int wi = t % 66;
        const int p  = t / 66;
        const int kd = p / 3, kh = p % 3;
        const int zd = dd + kd - 1, zh = h + kh - 1;
        float v = 0.0f;
        if ((unsigned)zd < DDIM && (unsigned)zh < HH && wi < 64) v = xsf[p][ci][wi];
        xsbT[p][wi][ci] = (__bf16)v;
    }
    __syncthreads();

    // ---- main MAC loop: branch-free, fully unrolled ----
    v8f acc0 = {}; v8f acc1 = {};
    const __bf16* wA = wfragA + ((size_t)mi * 32 + lane) * 16;

    #pragma unroll
    for (int p = 0; p < 9; ++p) {
        #pragma unroll
        for (int kc = 0; kc < 2; ++kc) {
            const v16bf a = *(const v16bf*)(wA + (p * 2 + kc) * 2048);
            int wi0, wi1;
            if (kc == 0) {                 // lanes lo: kw=0 (w-1); lanes hi: kw=1 (w)
                wi0 = n0 + hi - 1; if (wi0 < 0) wi0 = 65;   // left halo -> zero row
                wi1 = n1 + hi - 1;
            } else {                       // lanes lo: kw=2 (w+1, slot 64 = zero pad); hi: K>=48 pad
                wi0 = hi ? 65 : (n0 + 1);
                wi1 = hi ? 65 : (n1 + 1);
            }
            const v16bf b0 = *(const v16bf*)&xsbT[p][wi0][0];
            const v16bf b1 = *(const v16bf*)&xsbT[p][wi1][0];
            acc0 = __builtin_amdgcn_wmma_f32_16x16x32_bf16(false, a, false, b0, (short)0, acc0, false, false);
            acc1 = __builtin_amdgcn_wmma_f32_16x16x32_bf16(false, a, false, b1, (short)0, acc1, false, false);
        }
    }

    // conv (f32) into d_out; lif_kernel reads it and overwrites with spikes
    #pragma unroll
    for (int r = 0; r < 8; ++r) {
        const int c = mi * 16 + hi * 8 + r;
        const size_t base = ((((size_t)b * COUT + c) * DDIM + dd) * HH + h) * WW;
        out[base + n0] = acc0[r];
        out[base + n1] = acc1[r];
    }
}

// LIF recurrence. Membrane stays in WMMA C/D f32 fragments for all 32 steps.
__global__ __launch_bounds__(256) void lif_kernel(
    const __bf16* __restrict__ dfrag, const float* __restrict__ inv_norm,
    const float* __restrict__ bvec, const float* __restrict__ beta_p,
    float* __restrict__ io)
{
    __shared__ __bf16 spkT[WW][72];   // [pixel][channel], 144B row (16B aligned)

    const int blk = blockIdx.x;
    const int h   = blk & 63;
    const int b   = blk >> 6;

    const int wave   = threadIdx.x >> 5;
    const int lane   = threadIdx.x & 31;
    const int laneLo = lane & 15;
    const int hi     = lane >> 4;

    const int mi = wave >> 1;
    const int nj = (wave & 1) * 2;
    const int n0 = nj * 16 + laneLo;
    const int n1 = n0 + 16;
    const int cb = mi * 16 + hi * 8;   // first of 8 channels this lane owns

    const float beta = beta_p[0];
    const float omb  = 1.0f - beta;

    // pre-swizzled Gram-matrix A fragments (held in VGPRs all 32 steps)
    const v16bf dA0 = *(const v16bf*)(dfrag + ((size_t)(0 * 4 + mi) * 32 + lane) * 16);
    const v16bf dA1 = *(const v16bf*)(dfrag + ((size_t)(1 * 4 + mi) * 32 + lane) * 16);

    float inv8[8], b8[8];
    #pragma unroll
    for (int r = 0; r < 8; ++r) { inv8[r] = inv_norm[cb + r]; b8[r] = bvec[cb + r]; }

    // spk_0 = 0 (vectorized clear)
    for (int i = threadIdx.x; i < (WW * 72 * 2) / 16; i += 256)
        ((uint4*)&spkT[0][0])[i] = make_uint4(0, 0, 0, 0);
    __syncthreads();

    v8f mem0 = {}; v8f mem1 = {};

    for (int t = 0; t < DDIM; ++t) {
        // B fragments of spk_{t-1}: contiguous 16-channel runs
        const v16bf s00 = *(const v16bf*)&spkT[n0][hi * 16];
        const v16bf s01 = *(const v16bf*)&spkT[n1][hi * 16];
        const v16bf s10 = *(const v16bf*)&spkT[n0][32 + hi * 16];
        const v16bf s11 = *(const v16bf*)&spkT[n1][32 + hi * 16];

        v8f rst0 = {}, rst1 = {};
        rst0 = __builtin_amdgcn_wmma_f32_16x16x32_bf16(false, dA0, false, s00, (short)0, rst0, false, false);
        rst0 = __builtin_amdgcn_wmma_f32_16x16x32_bf16(false, dA1, false, s10, (short)0, rst0, false, false);
        rst1 = __builtin_amdgcn_wmma_f32_16x16x32_bf16(false, dA0, false, s01, (short)0, rst1, false, false);
        rst1 = __builtin_amdgcn_wmma_f32_16x16x32_bf16(false, dA1, false, s11, (short)0, rst1, false, false);

        __syncthreads();   // everyone done reading spk_{t-1}

        union { __bf16 hx[8]; uint4 u; } pk0, pk1;
        #pragma unroll
        for (int r = 0; r < 8; ++r) {
            const int c = cb + r;
            const size_t idx = ((((size_t)b * COUT + c) * DDIM + t) * HH + h) * WW;
            if (t + 1 < DDIM && r == 0) {           // hide next step's strided reads
                __builtin_prefetch(io + idx + (size_t)HH * WW + n0, 0, 0);
                __builtin_prefetch(io + idx + (size_t)HH * WW + n1, 0, 0);
            }
            const float conv0 = io[idx + n0];
            const float conv1 = io[idx + n1];
            const float m0 = (mem0[r] - rst0[r]) * beta + conv0 * omb;
            const float m1 = (mem1[r] - rst1[r]) * beta + conv1 * omb;
            mem0[r] = m0; mem1[r] = m1;
            const float sp0 = (m0 * inv8[r] - b8[r]) > 0.0f ? 1.0f : 0.0f;
            const float sp1 = (m1 * inv8[r] - b8[r]) > 0.0f ? 1.0f : 0.0f;
            io[idx + n0] = sp0;                      // overwrite conv with spike (same element)
            io[idx + n1] = sp1;
            pk0.hx[r] = (__bf16)sp0;
            pk1.hx[r] = (__bf16)sp1;
        }
        *(uint4*)((char*)&spkT[n0][0] + cb * 2) = pk0.u;   // 16B-aligned packed spike store
        *(uint4*)((char*)&spkT[n1][0] + cb * 2) = pk1.u;

        __syncthreads();   // spk_t visible before next step's reads
    }
}

extern "C" void kernel_launch(void* const* d_in, const int* in_sizes, int n_in,
                              void* d_out, int out_size, void* d_ws, size_t ws_size,
                              hipStream_t stream) {
    (void)in_sizes; (void)n_in; (void)out_size; (void)ws_size;
    const float* x    = (const float*)d_in[0];
    const float* w    = (const float*)d_in[1];
    const float* beta = (const float*)d_in[2];
    const float* bvec = (const float*)d_in[3];
    float* out = (float*)d_out;

    char* ws = (char*)d_ws;
    float*  inv_norm = (float*)(ws);
    __bf16* dfrag    = (__bf16*)(ws + 256);
    __bf16* wfragA   = (__bf16*)(ws + 8448);

    prep_kernel<<<64, 256, 0, stream>>>(w, inv_norm, dfrag, wfragA);
    conv3d_kernel<<<BB * DDIM * HH, 256, 0, stream>>>(x, wfragA, out);
    lif_kernel<<<BB * HH, 256, 0, stream>>>(dfrag, inv_norm, bvec, beta, out);
}
